// SlayerSNN_20538533609769
// MI455X (gfx1250) — matrix operand order repaired and checked by reading, not compiled, not run on hardware
//
#include <hip/hip_runtime.h>

// ---------------------------------------------------------------------------
// SlayerSNN forward for MI455X (gfx1250, wave32).
//   gemm1_embed : h = W1 * gather(emb, feed)     (bf16 WMMA, f32 accum)
//   conv_spike  : u = causal-conv(h, srm) ; s = spike-scan(u)   (in-place)
//   gemm2       : o = W2 * s                      (bf16 WMMA, M padded to 16)
//   conv_spike  : out = spike-scan(conv(o))
// Workspace: h (64*512*256 f32) + opad (64*16*256 f32) = ~34.6 MB.
// ---------------------------------------------------------------------------

typedef __attribute__((ext_vector_type(16))) __bf16 v16bf;
typedef __attribute__((ext_vector_type(8)))  __bf16 v8bf;
typedef __attribute__((ext_vector_type(4)))  __bf16 v4bf;
typedef __attribute__((ext_vector_type(8)))  float  v8f;
typedef __attribute__((ext_vector_type(4)))  float  v4f;

static constexpr int   BATCH = 64;
static constexpr int   TLEN  = 256;
static constexpr int   EMB   = 300;    // multiple of 4 -> float4 staging
static constexpr int   HID   = 512;
static constexpr int   OUTC  = 10;
static constexpr int   OUTP  = 16;     // padded M for GEMM2
static constexpr int   LDSK  = 40;     // padded LDS row stride (bf16 elems)
static constexpr float THETA = 10.0f;
static constexpr float TS    = 1.0f;

static __device__ __forceinline__ v16bf cat8(v8bf lo, v8bf hi) {
  return __builtin_shufflevector(lo, hi, 0, 1, 2, 3, 4, 5, 6, 7,
                                         8, 9, 10, 11, 12, 13, 14, 15);
}

static __device__ __forceinline__ v4bf pack4(v4f v) {
  v4bf r;
  r[0] = (__bf16)v[0]; r[1] = (__bf16)v[1];
  r[2] = (__bf16)v[2]; r[3] = (__bf16)v[3];
  return r;
}

// LDS byte offset of a generic pointer to __shared__ (ISA §10.2: addr[31:0]).
static __device__ __forceinline__ unsigned lds_off(const void* p) {
  return (unsigned)(unsigned long long)p;
}

// ---------------------------------------------------------------------------
// GEMM1 fused with embedding gather.
// grid = (BATCH * TLEN/128, HID/128), block = 256 (8 waves).
// Block tile: 128 (M) x 128 (N=t within one batch).  K = EMB padded to 320.
// ---------------------------------------------------------------------------
__global__ __launch_bounds__(256)
void gemm1_embed(const int* __restrict__ feed, const float* __restrict__ emb,
                 const float* __restrict__ W1, float* __restrict__ h)
{
  __shared__ __bf16 As[128 * LDSK];
  __shared__ __bf16 Bs[128 * LDSK];

  const int tid  = threadIdx.x;
  const int lane = tid & 31;
  const int wave = tid >> 5;
  const int b    = blockIdx.x >> 1;
  const int tb0  = (blockIdx.x & 1) * 128;
  const int mb0  = blockIdx.y * 128;

  // staging ownership: thread -> (row srow, 16-wide half shalf)
  const int srow  = tid >> 1;
  const int shalf = (tid & 1) * 16;
  const long grow = (long)feed[b * TLEN + tb0 + srow];      // gather index (loop-invariant)
  const float* __restrict__ embrow = emb + grow * EMB;
  const float* __restrict__ w1row  = W1 + (long)(mb0 + srow) * EMB;

  // One-shot prefetch: the gathered emb row (random in a 60 MB table) and the
  // W1 panel row are both reused across all 10 k-steps.
  __builtin_prefetch(embrow, 0, 3);
  __builtin_prefetch(w1row, 0, 3);

  // wave tile: 32 (M) x 64 (N)  -> 2 x 4 WMMA fragments
  const int wm = (wave & 3) * 32;
  const int wn = (wave >> 2) * 64;

  v8f acc[2][4];
#pragma unroll
  for (int mi = 0; mi < 2; ++mi)
#pragma unroll
    for (int ni = 0; ni < 4; ++ni)
#pragma unroll
      for (int e = 0; e < 8; ++e) acc[mi][ni][e] = 0.0f;

  const int halfsel = lane >> 4;
  const int l15     = lane & 15;

  for (int kk = 0; kk < EMB; kk += 32) {
    __syncthreads();                       // protect previous-iter frag reads
    // float4-granularity staging; EMB % 4 == 0 so a vector never straddles.
#pragma unroll
    for (int j = 0; j < 16; j += 4) {
      const int gk = kk + shalf + j;
      v4f va = {0.0f, 0.0f, 0.0f, 0.0f};
      v4f vb = {0.0f, 0.0f, 0.0f, 0.0f};
      if (gk + 4 <= EMB) {
        va = *(const v4f*)(w1row + gk);
        vb = *(const v4f*)(embrow + gk);
      }
      *(v4bf*)&As[srow * LDSK + shalf + j] = pack4(va);
      *(v4bf*)&Bs[srow * LDSK + shalf + j] = pack4(vb);
    }
    __syncthreads();

    v16bf afrag[2], bfrag[4];
#pragma unroll
    for (int mi = 0; mi < 2; ++mi) {
      const __bf16* p = &As[(wm + mi * 16 + l15) * LDSK + halfsel * 8];
      afrag[mi] = cat8(*(const v8bf*)p, *(const v8bf*)(p + 16));
    }
#pragma unroll
    for (int ni = 0; ni < 4; ++ni) {
      const __bf16* p = &Bs[(wn + ni * 16 + l15) * LDSK + halfsel * 8];
      bfrag[ni] = cat8(*(const v8bf*)p, *(const v8bf*)(p + 16));
    }
#pragma unroll
    for (int mi = 0; mi < 2; ++mi)
#pragma unroll
      for (int ni = 0; ni < 4; ++ni)
        acc[mi][ni] = __builtin_amdgcn_wmma_f32_16x16x32_bf16(
            false, afrag[mi], false, bfrag[ni], (short)0, acc[mi][ni],
            false, false);
  }

  const int rofs = halfsel * 8;
#pragma unroll
  for (int mi = 0; mi < 2; ++mi)
#pragma unroll
    for (int ni = 0; ni < 4; ++ni)
#pragma unroll
      for (int r = 0; r < 8; ++r) {
        const int m = mb0 + wm + mi * 16 + rofs + r;
        const int t = tb0 + wn + ni * 16 + l15;
        h[((long)b * HID + m) * TLEN + t] = acc[mi][ni][r];
      }
}

// ---------------------------------------------------------------------------
// GEMM2: o = W2 (10x512, padded to 16) * s (per-batch 512x256 spikes).
// grid = BATCH, block = 256 (8 waves); wave w owns cols [32w, 32w+32).
// ---------------------------------------------------------------------------
__global__ __launch_bounds__(256)
void gemm2(const float* __restrict__ s, const float* __restrict__ W2,
           float* __restrict__ opad)
{
  __shared__ __bf16 As[OUTP * LDSK];
  __shared__ __bf16 Bs[TLEN * LDSK];

  const int tid  = threadIdx.x;
  const int lane = tid & 31;
  const int wave = tid >> 5;
  const int b    = blockIdx.x;

  v8f acc[2];
#pragma unroll
  for (int ni = 0; ni < 2; ++ni)
#pragma unroll
    for (int e = 0; e < 8; ++e) acc[ni][e] = 0.0f;

  const int halfsel = lane >> 4;
  const int l15     = lane & 15;

  for (int kk = 0; kk < HID; kk += 32) {
    __syncthreads();
    // stage A: 16 rows x 32 k (rows >= OUTC zero-padded), float4 granularity
    if (tid < 32) {
      const int m    = tid >> 1;
      const int half = (tid & 1) * 16;
#pragma unroll
      for (int j = 0; j < 16; j += 4) {
        v4f v = {0.0f, 0.0f, 0.0f, 0.0f};
        if (m < OUTC) v = *(const v4f*)(W2 + m * HID + kk + half + j);
        *(v4bf*)&As[m * LDSK + half + j] = pack4(v);
      }
    }
    // stage B: Bs[n][k] = s[b][kk+k][n]; thread owns one k, 32 consecutive n.
    // Global side vectorized (b128); LDS side is k-strided b16 by necessity.
    {
      const int k  = tid & 31;
      const int nb = (tid >> 5) * 32;
      const float* __restrict__ sp = s + ((long)b * HID + kk + k) * TLEN + nb;
#pragma unroll
      for (int jj = 0; jj < 32; jj += 4) {
        const v4f q = *(const v4f*)(sp + jj);
        Bs[(nb + jj + 0) * LDSK + k] = (__bf16)q[0];
        Bs[(nb + jj + 1) * LDSK + k] = (__bf16)q[1];
        Bs[(nb + jj + 2) * LDSK + k] = (__bf16)q[2];
        Bs[(nb + jj + 3) * LDSK + k] = (__bf16)q[3];
      }
    }
    __syncthreads();

    v16bf af, bfrag[2];
    {
      const __bf16* p = &As[l15 * LDSK + halfsel * 8];
      af = cat8(*(const v8bf*)p, *(const v8bf*)(p + 16));
    }
#pragma unroll
    for (int ni = 0; ni < 2; ++ni) {
      const __bf16* p = &Bs[(wave * 32 + ni * 16 + l15) * LDSK + halfsel * 8];
      bfrag[ni] = cat8(*(const v8bf*)p, *(const v8bf*)(p + 16));
    }
#pragma unroll
    for (int ni = 0; ni < 2; ++ni)
      acc[ni] = __builtin_amdgcn_wmma_f32_16x16x32_bf16(
          false, af, false, bfrag[ni], (short)0, acc[ni], false, false);
  }

  const int rofs = halfsel * 8;
#pragma unroll
  for (int ni = 0; ni < 2; ++ni)
#pragma unroll
    for (int r = 0; r < 8; ++r) {
      const int m = rofs + r;
      const int t = wave * 32 + ni * 16 + l15;
      opad[((long)b * OUTP + m) * TLEN + t] = acc[ni][r];
    }
}

// ---------------------------------------------------------------------------
// Fused psp (causal 77-tap conv along T) + spike scan (10-deep refractory).
// One wave per (b,c) row; block = 8 waves. Row stage-in and spike store-out
// are pure global<->LDS copies -> CDNA5 async LDS ops (ASYNCcnt-tracked,
// no VGPR round-trip). Lanes do the conv in parallel, lane 0 runs the
// strictly-sequential scan, async b128 stores write the row back.
// Safe in-place (out may alias in): each row fully staged to LDS first.
// Row mapping: inrow = (row / cg_out) * cg_in + (row % cg_out).
// ---------------------------------------------------------------------------
__global__ __launch_bounds__(256)
void conv_spike(const float* __restrict__ in, float* __restrict__ out,
                const float* __restrict__ srm, int K,
                const float* __restrict__ refk, int R,
                int nrows, int cg_out, int cg_in)
{
  __shared__ float ker[128];
  __shared__ float rowbuf[8][TLEN];
  __shared__ float ubuf[8][TLEN];

  const int tid  = threadIdx.x;
  const int lane = tid & 31;
  const int wave = tid >> 5;

  for (int i = tid; i < 128; i += 256) ker[i] = (i < K) ? srm[i] : 0.0f;

  const int  row    = blockIdx.x * 8 + wave;
  const bool active = (row < nrows);
  const int  inrow  = (row / cg_out) * cg_in + (row % cg_out);

  if (active) {
    const float* __restrict__ ip = in + (long)inrow * TLEN;
    // 2 x b128 async copies per lane cover the 1 KB row (coalesced per wave).
#pragma unroll
    for (int c = 0; c < 2; ++c) {
      const int i = c * 128 + lane * 4;
      const unsigned loff = lds_off(&rowbuf[wave][i]);
      const float* gp = ip + i;
      asm volatile("global_load_async_to_lds_b128 %0, %1, off"
                   :: "v"(loff), "v"(gp) : "memory");
    }
  }
  asm volatile("s_wait_asynccnt 0x0" ::: "memory");
  __syncthreads();

  if (active) {
    for (int i = lane; i < TLEN; i += 32) {
      float a = 0.0f;
      const int kmax = (K - 1 < i) ? (K - 1) : i;
      for (int k = 0; k <= kmax; ++k) a += ker[k] * rowbuf[wave][i - k];
      ubuf[wave][i] = a * TS;
    }
  }
  __syncthreads();

  if (active && lane == 0) {
    float fut[12], r[12];
#pragma unroll
    for (int j = 0; j < 12; ++j) {
      fut[j] = (j < R) ? refk[j + 1] : 0.0f;
      r[j]   = 0.0f;
    }
    for (int t = 0; t < TLEN; ++t) {
      const float m = ubuf[wave][t] + r[0];
      const float sv = (m >= THETA) ? 1.0f : 0.0f;
#pragma unroll
      for (int j = 0; j < 11; ++j) r[j] = r[j + 1];
      r[11] = 0.0f;
#pragma unroll
      for (int j = 0; j < 12; ++j) r[j] += sv * fut[j];
      ubuf[wave][t] = sv;            // reuse u row as spike row
    }
  }
  __syncthreads();

  if (active) {
    float* __restrict__ op = out + (long)row * TLEN;
#pragma unroll
    for (int c = 0; c < 2; ++c) {
      const int i = c * 128 + lane * 4;
      const unsigned loff = lds_off(&ubuf[wave][i]);
      float* gp = op + i;
      asm volatile("global_store_async_from_lds_b128 %0, %1, off"
                   :: "v"(gp), "v"(loff) : "memory");
    }
  }
  // S_ENDPGM performs an implicit wait-idle; explicit wait keeps LDS reuse
  // safe if the compiler ever tail-merges.
  asm volatile("s_wait_asynccnt 0x0" ::: "memory");
}

// ---------------------------------------------------------------------------
extern "C" void kernel_launch(void* const* d_in, const int* in_sizes, int n_in,
                              void* d_out, int out_size, void* d_ws, size_t ws_size,
                              hipStream_t stream) {
  const int*   feed = (const int*)d_in[0];
  const float* emb  = (const float*)d_in[1];
  const float* W1   = (const float*)d_in[2];
  const float* W2   = (const float*)d_in[3];
  const float* srm  = (const float*)d_in[4];
  const float* refk = (const float*)d_in[5];
  const int K = in_sizes[4];          // srm kernel length (77)
  const int R = in_sizes[5] - 1;      // refractory future length (10)

  float* out  = (float*)d_out;
  float* h    = (float*)d_ws;                          // 64*512*256 f32
  float* opad = h + (size_t)BATCH * HID * TLEN;        // 64*16*256  f32

  // 1) h = W1 * gather(emb, feed)
  dim3 g1(BATCH * (TLEN / 128), HID / 128);
  gemm1_embed<<<g1, 256, 0, stream>>>(feed, emb, W1, h);

  // 2) spikes layer 1 (in-place over h)
  conv_spike<<<(BATCH * HID) / 8, 256, 0, stream>>>(h, h, srm, K, refk, R,
                                                    BATCH * HID, HID, HID);

  // 3) o = W2 * s   (M padded to 16)
  gemm2<<<BATCH, 256, 0, stream>>>(h, W2, opad);

  // 4) spikes layer 2 -> output
  conv_spike<<<(BATCH * OUTC + 7) / 8, 256, 0, stream>>>(opad, out, srm, K, refk, R,
                                                         BATCH * OUTC, OUTC, OUTP);
}
